// GeneralizedSupervisedNTXenLoss_61813169324825
// MI455X (gfx1250) — matrix operand order repaired and checked by reading, not compiled, not run on hardware
//
#include <hip/hip_runtime.h>
#include <hip/hip_bf16.h>

typedef __attribute__((ext_vector_type(16))) _Float16 v16h;
typedef __attribute__((ext_vector_type(8)))  float    v8f;

#define NPTS   4096
#define TWO_N  8192
#define DDIM   128
#define INV_T  10.0f
#define LOG2E  1.4426950408889634f
#define LN2    0.6931471805599453f
#define G1     0.02f      /* 1/(2*5^2)  */
#define CATSIM 0.5f
#define BM     32
#define BN     64
#define APAD   8
#define ASTR   (DDIM + APAD)   /* 136 halves, 272B row stride (16B aligned) */

// ---------------------------------------------------------------- prep rows
__global__ void ntxen_prep_rows(const float* __restrict__ zi,
                                const float* __restrict__ zj,
                                _Float16* __restrict__ uhi,
                                _Float16* __restrict__ ulo) {
  int r = blockIdx.x;              // 0..8191
  int t = threadIdx.x;             // 0..127
  const float* src = (r < NPTS) ? (zi + (size_t)r * DDIM)
                                : (zj + (size_t)(r - NPTS) * DDIM);
  float v = src[t];
  __shared__ float red[DDIM];
  red[t] = v * v;
  __syncthreads();
  for (int s = DDIM / 2; s > 0; s >>= 1) {
    if (t < s) red[t] += red[t + s];
    __syncthreads();
  }
  float x = v * rsqrtf(red[0]);
  _Float16 hi = (_Float16)x;
  _Float16 lo = (_Float16)(x - (float)hi);
  uhi[(size_t)r * DDIM + t] = hi;
  ulo[(size_t)r * DDIM + t] = lo;
}

// ------------------------------------------------------------- label prep
// y tiled => rowsum over 2N = 2*(sum over N) - self term (diag removed).
__global__ void ntxen_prep_labels(const float* __restrict__ labels,
                                  float* __restrict__ y1, float* __restrict__ y2,
                                  float* __restrict__ rs1, float* __restrict__ rs2) {
  __shared__ float sc[NPTS];
  __shared__ float sk[NPTS];
  int t = threadIdx.x;
  for (int n = t; n < NPTS; n += blockDim.x) {
    sc[n] = labels[n * 2 + 0];
    sk[n] = labels[n * 2 + 1];
  }
  __syncthreads();
  int r = blockIdx.x * blockDim.x + t;     // 0..8191
  float a = sc[r & (NPTS - 1)];
  float b = sk[r & (NPTS - 1)];
  float s1 = 0.0f;
  int cnt = 0;
  for (int n = 0; n < NPTS; ++n) {
    float d = a - sc[n];
    s1 += __builtin_amdgcn_exp2f(-(G1 * LOG2E) * d * d);  // exponent >= -104
    cnt += (sk[n] == b) ? 1 : 0;
  }
  float rowsum1 = 2.0f * s1 - 1.0f;                 // remove j==r (exp(0)=1)
  float cnt2 = 2.0f * (float)cnt - 1.0f;            // equal count, j != r
  float rowsum2 = cnt2 + CATSIM * ((float)(TWO_N - 1) - cnt2);
  y1[r] = a; y2[r] = b; rs1[r] = rowsum1; rs2[r] = rowsum2;
}

// ---------------------------------------------------------- async staging
__device__ __forceinline__ void issue_async_b128(_Float16* lds_dst,
                                                 const _Float16* gsrc) {
  // CDNA5 async copy: memory -> LDS, tracked by ASYNCcnt, no staging VGPRs.
  uint32_t loff = (uint32_t)(uintptr_t)lds_dst;   // low 32 bits = LDS offset
  asm volatile("global_load_async_to_lds_b128 %0, %1, off"
               :: "v"(loff), "v"(gsrc)
               : "memory");
}

__device__ __forceinline__ void wait_async0() {
  asm volatile("s_wait_asynccnt 0x0" ::: "memory");
}

__device__ __forceinline__ void stage_B_async(const _Float16* __restrict__ uhi,
                                              const _Float16* __restrict__ ulo,
                                              _Float16* __restrict__ bhi,
                                              _Float16* __restrict__ blo,
                                              int ct, int t) {
  for (int j = 0; j < 4; ++j) {
    int e = t * 32 + j * 8;
    int row = e >> 7, k = e & 127;
    size_t g = (size_t)(ct * BN + row) * DDIM + k;
    issue_async_b128(&bhi[row * ASTR + k], uhi + g);
    issue_async_b128(&blo[row * ASTR + k], ulo + g);
  }
}

// -------------------------------------------------- fused GEMM + loss epilogue
__launch_bounds__(256)
__global__ void ntxen_main(const _Float16* __restrict__ uhi,
                           const _Float16* __restrict__ ulo,
                           const float* __restrict__ y1, const float* __restrict__ y2,
                           const float* __restrict__ rs1, const float* __restrict__ rs2,
                           float* __restrict__ out) {
  __shared__ __align__(16) _Float16 sAhi[BM * ASTR];
  __shared__ __align__(16) _Float16 sAlo[BM * ASTR];
  __shared__ __align__(16) _Float16 sBhi[2][BN * ASTR];   // double buffered
  __shared__ __align__(16) _Float16 sBlo[2][BN * ASTR];
  __shared__ float rowS[BM], rowW1[BM], rowW2[BM];

  const int t        = threadIdx.x;
  const int lane     = t & 31;
  const int wave     = t >> 5;        // 0..7
  const int subRow   = wave & 1;      // which 16-row sub-tile of BM=32
  const int subCol   = wave >> 1;     // which 16-col sub-tile of BN=64
  const int laneHalf = lane >> 4;     // 0 or 1
  const int ln       = lane & 15;
  const int rowBase  = blockIdx.x * BM;

  if (t < BM) { rowS[t] = 0.0f; rowW1[t] = 0.0f; rowW2[t] = 0.0f; }

  // ---- stage A block (rows rowBase..rowBase+31) into LDS
  for (int j = 0; j < 2; ++j) {
    int e = t * 16 + j * 8;
    int row = e >> 7, k = e & 127;
    size_t g = (size_t)(rowBase + row) * DDIM + k;
    *reinterpret_cast<uint4*>(&sAhi[row * ASTR + k]) =
        *reinterpret_cast<const uint4*>(uhi + g);
    *reinterpret_cast<uint4*>(&sAlo[row * ASTR + k]) =
        *reinterpret_cast<const uint4*>(ulo + g);
  }
  // prefetch first B tile while A settles
  stage_B_async(uhi, ulo, sBhi[0], sBlo[0], 0, t);
  __syncthreads();

  // ---- hoist A fragments into registers (reused by every column tile)
  // 16-bit A 16x32 layout: lane = 16*laneHalf + m; element e -> K =
  // laneHalf*8 + (e&7) + (e<8 ? 0 : 16)
  v16h aHi[4], aLo[4];
  {
    int m = subRow * 16 + ln;
    for (int ks = 0; ks < 4; ++ks) {
      int kb = ks * 32 + laneHalf * 8;
      union { v16h v; uint4 q[2]; } uh, ul;
      uh.q[0] = *reinterpret_cast<const uint4*>(&sAhi[m * ASTR + kb]);
      uh.q[1] = *reinterpret_cast<const uint4*>(&sAhi[m * ASTR + kb + 16]);
      ul.q[0] = *reinterpret_cast<const uint4*>(&sAlo[m * ASTR + kb]);
      ul.q[1] = *reinterpret_cast<const uint4*>(&sAlo[m * ASTR + kb + 16]);
      aHi[ks] = uh.v; aLo[ks] = ul.v;
    }
  }

  // per-row label values for the 8 C rows this lane owns
  float ry1[8], ry2[8];
  for (int r = 0; r < 8; ++r) {
    int gi = rowBase + subRow * 16 + laneHalf * 8 + r;
    ry1[r] = y1[gi]; ry2[r] = y2[gi];
  }

  float accS[8], accW1[8], accW2[8];
  for (int r = 0; r < 8; ++r) { accS[r] = 0.0f; accW1[r] = 0.0f; accW2[r] = 0.0f; }

  const int numTiles = TWO_N / BN;   // 128
  for (int ct = 0; ct < numTiles; ++ct) {
    const int cur = ct & 1;
    wait_async0();         // my async writes into buf[cur] have landed
    __syncthreads();       // everyone's have; all reads of buf[cur^1] done
    if (ct + 1 < numTiles)
      stage_B_async(uhi, ulo, sBhi[cur ^ 1], sBlo[cur ^ 1], ct + 1, t);

    // column-tile labels (issue early so latency hides under WMMA)
    int   gj  = ct * BN + subCol * 16 + ln;
    float cy1 = y1[gj], cy2 = y2[gj];

    // split-f16 GEMM: 3 independent accumulator chains (hi*hi, hi*lo, lo*hi)
    v8f c0 = {}, c1 = {}, c2 = {};
    for (int ks = 0; ks < 4; ++ks) {
      // 16-bit B 32x16 layout: lane = 16*laneHalf + n; element e -> K = laneHalf*8 + (e&7) + (e<8 ? 0 : 16)
      int n = subCol * 16 + ln;
      int kb = ks * 32 + laneHalf * 8;
      union { v16h v; uint4 q[2]; } bh, bl;
      const uint4* ph = reinterpret_cast<const uint4*>(&sBhi[cur][n * ASTR + kb]);
      const uint4* pl = reinterpret_cast<const uint4*>(&sBlo[cur][n * ASTR + kb]);
      bh.q[0] = ph[0]; bh.q[1] = ph[2];
      bl.q[0] = pl[0]; bl.q[1] = pl[2];
      c0 = __builtin_amdgcn_wmma_f32_16x16x32_f16(false, aHi[ks], false, bh.v,
                                                  (short)0, c0, false, false);
      c1 = __builtin_amdgcn_wmma_f32_16x16x32_f16(false, aHi[ks], false, bl.v,
                                                  (short)0, c1, false, false);
      c2 = __builtin_amdgcn_wmma_f32_16x16x32_f16(false, aLo[ks], false, bh.v,
                                                  (short)0, c2, false, false);
    }

    // fused epilogue: C element r -> (M = laneHalf*8 + r, N = ln); branch-free
    for (int r = 0; r < 8; ++r) {
      int   gi   = rowBase + subRow * 16 + laneHalf * 8 + r;
      float mask = (gi != gj) ? 1.0f : 0.0f;        // -INF diagonal: drop term
      float sim  = (c0[r] + c1[r] + c2[r]) * INV_T; // |sim| <= 10
      accS[r]  += mask * __builtin_amdgcn_exp2f(sim * LOG2E);
      float dy  = ry1[r] - cy1;
      accW1[r] += mask * __builtin_amdgcn_exp2f(-(G1 * LOG2E) * dy * dy) * sim;
      accW2[r] += mask * ((ry2[r] == cy2) ? 1.0f : CATSIM) * sim;
    }
  }

  // reduce the 16 lanes sharing each row, then merge the 4 column-waves in LDS
  for (int r = 0; r < 8; ++r) {
    float s = accS[r], w1 = accW1[r], w2 = accW2[r];
    for (int off = 1; off < 16; off <<= 1) {
      s  += __shfl_xor(s,  off, 32);
      w1 += __shfl_xor(w1, off, 32);
      w2 += __shfl_xor(w2, off, 32);
    }
    if (ln == 0) {
      int lr = subRow * 16 + laneHalf * 8 + r;
      atomicAdd(&rowS[lr],  s);
      atomicAdd(&rowW1[lr], w1);
      atomicAdd(&rowW2[lr], w2);
    }
  }
  __syncthreads();

  if (t < BM) {
    int gi = rowBase + t;
    float term = rowW1[t] / rs1[gi] + rowW2[t] / rs2[gi]
               - 2.0f * (LN2 * __builtin_amdgcn_logf(rowS[t]));
    atomicAdd(out, -term * (1.0f / (float)NPTS));
  }
}

// ---------------------------------------------------------------- launcher
extern "C" void kernel_launch(void* const* d_in, const int* in_sizes, int n_in,
                              void* d_out, int out_size, void* d_ws, size_t ws_size,
                              hipStream_t stream) {
  (void)in_sizes; (void)n_in; (void)out_size; (void)ws_size;
  const float* zi     = (const float*)d_in[0];
  const float* zj     = (const float*)d_in[1];
  const float* labels = (const float*)d_in[2];
  float* out = (float*)d_out;

  char* ws = (char*)d_ws;
  _Float16* uhi = (_Float16*)ws;                                   // 2 MB
  _Float16* ulo = (_Float16*)(ws + (size_t)TWO_N * DDIM * 2);      // 2 MB
  float* y1  = (float*)(ws + (size_t)TWO_N * DDIM * 4);
  float* y2  = y1 + TWO_N;
  float* rs1 = y2 + TWO_N;
  float* rs2 = rs1 + TWO_N;

  hipMemsetAsync(d_out, 0, sizeof(float), stream);
  ntxen_prep_rows<<<TWO_N, DDIM, 0, stream>>>(zi, zj, uhi, ulo);
  ntxen_prep_labels<<<TWO_N / 256, 256, 0, stream>>>(labels, y1, y2, rs1, rs2);
  ntxen_main<<<TWO_N / BM, 256, 0, stream>>>(uhi, ulo, y1, y2, rs1, rs2, out);
}
